// MultiHeadAttention_31181462569562
// MI455X (gfx1250) — matrix operand order repaired and checked by reading, not compiled, not run on hardware
//
#include <hip/hip_runtime.h>

#define D_MODEL 1024
#define HEADS   16
#define DK      64
#define SEQ     2048
#define BATCH   4

#ifndef __has_builtin
#define __has_builtin(x) 0
#endif

#if __has_builtin(__builtin_amdgcn_global_load_async_to_lds_b128)
#define HAVE_ASYNC_LDS 1
#else
#define HAVE_ASYNC_LDS 0
#endif

typedef __attribute__((ext_vector_type(16))) __bf16 v16bf;
typedef __attribute__((ext_vector_type(8)))  float  v8f;

union Frag16 {            // 16 bf16 = one WMMA A/B operand (8 VGPRs)
    v16bf v;
    uint4 q[2];
};

__device__ __forceinline__ unsigned short f2bf(float f) {
    union { float f; unsigned int u; } x; x.f = f;
    unsigned int u = x.u + 0x7FFFu + ((x.u >> 16) & 1u);   // round-to-nearest-even
    return (unsigned short)(u >> 16);
}

__device__ __forceinline__ unsigned long long pack4bf(float4 s) {
    return (unsigned long long)f2bf(s.x)
         | ((unsigned long long)f2bf(s.y) << 16)
         | ((unsigned long long)f2bf(s.z) << 32)
         | ((unsigned long long)f2bf(s.w) << 48);
}

__device__ __forceinline__ v8f zero8() {
    v8f z; for (int i = 0; i < 8; ++i) z[i] = 0.0f; return z;
}

__device__ __forceinline__ v8f bf_wmma(const Frag16& a, const Frag16& b, v8f c) {
    // D = A(16x32 bf16) * B(32x16 bf16) + C(16x16 f32)
    return __builtin_amdgcn_wmma_f32_16x16x32_bf16(
        false, a.v, false, b.v, (short)0, c, false, false);
}

#if HAVE_ASYNC_LDS
// b128 async copy: param type per hipcc diagnostic is
// '__attribute__((__vector_size__(4*sizeof(int)))) int __device__ *' (AS1 v4i*)
typedef int v4i_ __attribute__((vector_size(16)));
__device__ __forceinline__ void async_ld_b128(const unsigned short* g, unsigned short* l) {
    __builtin_amdgcn_global_load_async_to_lds_b128(
        (__attribute__((address_space(1))) v4i_*)g,
        (__attribute__((address_space(3))) v4i_*)l, 0, 0);
}
__device__ __forceinline__ void wait_async0() {
#if __has_builtin(__builtin_amdgcn_s_wait_asynccnt)
    __builtin_amdgcn_s_wait_asynccnt(0);
#else
    asm volatile("s_wait_asynccnt 0x0" ::: "memory");
#endif
}
#endif

// ---------------------------------------------------------------------------
// Y[m][n] = sum_k X[m][k] * W[n][k]   (i.e. X @ W^T), fp32 in, WMMA bf16 core.
// MODE 0: out bf16 [B,H,S,DK], value scaled by `scale` (Q, scale=1/sqrt(dk))
// MODE 1: out bf16 [B,H,S,DK]                                   (K)
// MODE 2: out bf16 [B,H,DK,S]  (V transposed per head)          (V)
// MODE 3: out f32  [M,N] + bias[n]                              (final proj)
// Block: 128x64 output tile, 256 threads = 8 waves, wave tile 32x32, BK=64.
// ---------------------------------------------------------------------------
template <int MODE>
__global__ __launch_bounds__(256)
void gemm_xwt(const float* __restrict__ X, const float* __restrict__ W,
              const float* __restrict__ bias, void* __restrict__ Out,
              float scale)
{
    constexpr int BM = 128, BN = 64, BK = 64;
    constexpr int LAS = 72, LBS = 72;            // padded LDS strides (bf16 elems)
    __shared__ unsigned short sA[BM * LAS];
    __shared__ unsigned short sB[BN * LBS];

    const int tid  = threadIdx.x;
    const int lane = tid & 31;
    const int wave = tid >> 5;
    const int wm   = wave & 3;                   // 4 row groups of 32
    const int wn   = wave >> 2;                  // 2 col groups of 32
    const int half = lane >> 4;
    const int l16  = lane & 15;

    const int m0 = blockIdx.y * BM;
    const int n0 = blockIdx.x * BN;

    v8f acc[2][2];
    for (int i = 0; i < 2; ++i)
        for (int j = 0; j < 2; ++j) acc[i][j] = zero8();

    for (int kk = 0; kk < D_MODEL; kk += BK) {
        __syncthreads();
        // Stage A tile (128x64 f32 -> bf16): 2048 float4, 8 per thread.
        {
            int f4 = tid * 8;
            for (int u = 0; u < 8; ++u, ++f4) {
                int row = f4 >> 4;               // 16 float4 per row
                int c4  = (f4 & 15) * 4;
                float4 s = *(const float4*)(X + (size_t)(m0 + row) * D_MODEL + kk + c4);
                *(unsigned long long*)&sA[row * LAS + c4] = pack4bf(s);
            }
        }
        // Stage B tile (64x64 f32 -> bf16): 1024 float4, 4 per thread.
        {
            int f4 = tid * 4;
            for (int u = 0; u < 4; ++u, ++f4) {
                int row = f4 >> 4;
                int c4  = (f4 & 15) * 4;
                float4 s = *(const float4*)(W + (size_t)(n0 + row) * D_MODEL + kk + c4);
                *(unsigned long long*)&sB[row * LBS + c4] = pack4bf(s);
            }
        }
        if (kk + BK < D_MODEL) {                 // global_prefetch_b8 next A chunk
            __builtin_prefetch(X + (size_t)(m0 + (tid >> 1)) * D_MODEL + kk + BK, 0, 1);
        }
        __syncthreads();

        for (int c = 0; c < 2; ++c) {            // two K-chunks of 32
            Frag16 a[2], b[2];
            for (int i = 0; i < 2; ++i) {
                // A layout: lanes 0-15 row=l16 K{0-7,16-23}; lanes 16-31 K{8-15,24-31}
                const unsigned short* p =
                    &sA[(wm * 32 + i * 16 + l16) * LAS + c * 32 + half * 8];
                a[i].q[0] = *(const uint4*)(p);
                a[i].q[1] = *(const uint4*)(p + 16);
            }
            for (int j = 0; j < 2; ++j) {
                // B layout: lanes 0-15 col=l16 K{0-7,16-23}; lanes 16-31 K{8-15,24-31}
                const unsigned short* p =
                    &sB[(wn * 32 + j * 16 + l16) * LBS + c * 32 + half * 8];
                b[j].q[0] = *(const uint4*)(p);
                b[j].q[1] = *(const uint4*)(p + 16);
            }
            for (int i = 0; i < 2; ++i)
                for (int j = 0; j < 2; ++j)
                    acc[i][j] = bf_wmma(a[i], b[j], acc[i][j]);
        }
    }

    // C layout: element r -> row r (lanes 0-15) or row 8+r (lanes 16-31), col l16
    for (int i = 0; i < 2; ++i)
        for (int j = 0; j < 2; ++j) {
            int mbase = m0 + wm * 32 + i * 16 + (half ? 8 : 0);
            int n     = n0 + wn * 32 + j * 16 + l16;
            for (int r = 0; r < 8; ++r) {
                int m = mbase + r;
                float val = acc[i][j][r];
                if (MODE == 3) {
                    ((float*)Out)[(size_t)m * D_MODEL + n] = val + bias[n];
                } else {
                    int b_ = m >> 11, s_ = m & (SEQ - 1);
                    int h_ = n >> 6,  d_ = n & (DK - 1);
                    unsigned short o = f2bf(val * scale);
                    if (MODE == 2)
                        ((unsigned short*)Out)[(((size_t)b_ * HEADS + h_) * DK + d_) * SEQ + s_] = o;
                    else
                        ((unsigned short*)Out)[(((size_t)b_ * HEADS + h_) * SEQ + s_) * DK + d_] = o;
                }
            }
        }
}

// ---------------------------------------------------------------------------
// Flash attention: one block = 64 query rows of one (b,h); 4 waves x 16 rows.
// Q pre-scaled by 1/sqrt(dk).  K:[B,H,S,DK]  Vt:[B,H,DK,S]  (bf16)
// Out: concat fp32 [B,S,D_MODEL].
// K/V tiles staged with GLOBAL_LOAD_ASYNC_TO_LDS_B128 (ASYNCcnt) when available.
// ---------------------------------------------------------------------------
__global__ __launch_bounds__(128)
void attn_kernel(const unsigned short* __restrict__ Q,
                 const unsigned short* __restrict__ Kb,
                 const unsigned short* __restrict__ Vt,
                 float* __restrict__ Oc)
{
    constexpr int QT = 64, KT = 64;
    constexpr int LKS = 72, LVS = 72, LPS = 72;  // padded strides
    __shared__ unsigned short sK[KT * LKS];          // [key][d]
    __shared__ unsigned short sV[DK * LVS];          // [d][key]
    __shared__ unsigned short sP[4][16 * LPS];       // per-wave P tile [row][key]

    const int bh   = blockIdx.y;                     // b*16+h
    const int q0   = blockIdx.x * QT;
    const int tid  = threadIdx.x;
    const int lane = tid & 31;
    const int wave = tid >> 5;
    const int half = lane >> 4;
    const int l16  = lane & 15;

    const size_t baseQK = (size_t)bh * SEQ * DK;
    const size_t baseV  = (size_t)bh * DK * SEQ;

    // Persistent Q fragments for this wave's 16 rows (two K-chunks of 32).
    Frag16 qf[2];
    {
        const unsigned short* p = Q + baseQK + (size_t)(q0 + wave * 16 + l16) * DK;
        for (int c = 0; c < 2; ++c) {
            qf[c].q[0] = *(const uint4*)(p + c * 32 + half * 8);
            qf[c].q[1] = *(const uint4*)(p + c * 32 + half * 8 + 16);
        }
    }

    float mi[8], li[8];
    v8f   o[4];
    for (int r = 0; r < 8; ++r) { mi[r] = -1e30f; li[r] = 0.0f; }
    for (int t = 0; t < 4; ++t) o[t] = zero8();

    for (int kb = 0; kb < SEQ; kb += KT) {
        __syncthreads();
        // Stage K tile [64 keys][64 d] and V tile [64 d][64 keys] (bf16 -> LDS).
        for (int u = 0; u < 2; ++u) {
            int idx = tid * 2 + u;                   // 256 b128 copies (8 bf16 each), 16 apart
            int row = idx >> 2;
            int c   = (idx & 3) * 16;
            const unsigned short* gK = Kb + baseQK + (size_t)(kb + row) * DK + c;
            const unsigned short* gV = Vt + baseV + (size_t)row * SEQ + kb + c;
#if HAVE_ASYNC_LDS
            async_ld_b128(gK, &sK[row * LKS + c]);   // global_load_async_to_lds_b128
            async_ld_b128(gV, &sV[row * LVS + c]);
#else
            *(uint4*)&sK[row * LKS + c] = *(const uint4*)gK;
            *(uint4*)&sV[row * LVS + c] = *(const uint4*)gV;
#endif
        }
        if (kb + KT < SEQ) {                         // global_prefetch_b8 next block
            __builtin_prefetch(Kb + baseQK + (size_t)(kb + KT) * DK + tid * 32, 0, 1);
            __builtin_prefetch(Vt + baseV + (size_t)(tid >> 1) * SEQ + kb + KT, 0, 1);
        }
#if HAVE_ASYNC_LDS
        wait_async0();                               // s_wait_asynccnt 0
#endif
        __syncthreads();

        // Scores: 4 key tiles of 16, contraction over dk=64 (2 WMMAs each).
        v8f s[4];
        for (int t = 0; t < 4; ++t) {
            s[t] = zero8();
            for (int c = 0; c < 2; ++c) {
                Frag16 kf;
                const unsigned short* p = &sK[(t * 16 + l16) * LKS + c * 32 + half * 8];
                kf.q[0] = *(const uint4*)(p);
                kf.q[1] = *(const uint4*)(p + 16);
                s[t] = bf_wmma(qf[c], kf, s[t]);
            }
        }

        // Online softmax. Row r's 16 cols live in one 16-lane half -> xor reduce.
        float mnew[8], psum[8];
        for (int r = 0; r < 8; ++r) {
            float v = fmaxf(fmaxf(s[0][r], s[1][r]), fmaxf(s[2][r], s[3][r]));
            for (int off = 8; off >= 1; off >>= 1)
                v = fmaxf(v, __shfl_xor(v, off, 32));
            mnew[r] = fmaxf(mi[r], v);
            psum[r] = 0.0f;
        }
        for (int t = 0; t < 4; ++t)
            for (int r = 0; r < 8; ++r) {
                float p = __expf(s[t][r] - mnew[r]);
                psum[r] += p;
                int row = half ? 8 + r : r;
                sP[wave][row * LPS + t * 16 + l16] = f2bf(p);
            }
        for (int r = 0; r < 8; ++r) {
            float v = psum[r];
            for (int off = 8; off >= 1; off >>= 1)
                v += __shfl_xor(v, off, 32);
            float alpha = __expf(mi[r] - mnew[r]);
            li[r] = li[r] * alpha + v;
            mi[r] = mnew[r];
            for (int t = 0; t < 4; ++t) o[t][r] *= alpha;
        }

        // O += P(16x64) * V(64x64): A-frags from per-wave sP, B-frags from sV.
        // Same-wave LDS store->load is in-order; compiler inserts s_wait_dscnt.
        for (int c = 0; c < 2; ++c) {
            Frag16 pf;
            const unsigned short* pp = &sP[wave][l16 * LPS + c * 32 + half * 8];
            pf.q[0] = *(const uint4*)(pp);
            pf.q[1] = *(const uint4*)(pp + 16);
            for (int t = 0; t < 4; ++t) {
                Frag16 vf;
                const unsigned short* vp = &sV[(t * 16 + l16) * LVS + c * 32 + half * 8];
                vf.q[0] = *(const uint4*)(vp);
                vf.q[1] = *(const uint4*)(vp + 16);
                o[t] = bf_wmma(pf, vf, o[t]);
            }
        }
    }

    // Normalize and scatter into concat [B,S,D_MODEL].
    const int b_ = bh >> 4, h_ = bh & 15;
    for (int t = 0; t < 4; ++t)
        for (int r = 0; r < 8; ++r) {
            int row = q0 + wave * 16 + (half ? 8 : 0) + r;
            int d   = t * 16 + l16;
            Oc[((size_t)b_ * SEQ + row) * D_MODEL + h_ * DK + d] = o[t][r] / li[r];
        }
}

// ---------------------------------------------------------------------------
extern "C" void kernel_launch(void* const* d_in, const int* in_sizes, int n_in,
                              void* d_out, int out_size, void* d_ws, size_t ws_size,
                              hipStream_t stream)
{
    const float* q  = (const float*)d_in[0];
    const float* k  = (const float*)d_in[1];
    const float* v  = (const float*)d_in[2];
    const float* Wq = (const float*)d_in[3];
    const float* Wk = (const float*)d_in[5];
    const float* Wv = (const float*)d_in[7];
    const float* Wo = (const float*)d_in[9];
    const float* bo = (const float*)d_in[10];

    const size_t headElems = (size_t)BATCH * HEADS * SEQ * DK;   // 8,388,608
    unsigned short* Qb  = (unsigned short*)d_ws;
    unsigned short* Kbf = Qb  + headElems;
    unsigned short* Vtb = Kbf + headElems;
    float*          cat = (float*)(Vtb + headElems);             // [B,S,D_MODEL] f32

    dim3 ggrid(D_MODEL / 64, (BATCH * SEQ) / 128);               // 16 x 64
    dim3 gblk(256);

    gemm_xwt<0><<<ggrid, gblk, 0, stream>>>(q, Wq, nullptr, (void*)Qb,  0.125f);
    gemm_xwt<1><<<ggrid, gblk, 0, stream>>>(k, Wk, nullptr, (void*)Kbf, 1.0f);
    gemm_xwt<2><<<ggrid, gblk, 0, stream>>>(v, Wv, nullptr, (void*)Vtb, 1.0f);

    attn_kernel<<<dim3(SEQ / 64, BATCH * HEADS), 128, 0, stream>>>(Qb, Kbf, Vtb, cat);

    gemm_xwt<3><<<ggrid, gblk, 0, stream>>>(cat, Wo, bo, d_out, 1.0f);
}
